// ModelA_RWKV_20461224198648
// MI455X (gfx1250) — matrix operand order repaired and checked by reading, not compiled, not run on hardware
//
#include <hip/hip_runtime.h>
#include <cstdint>
#include <cstddef>

// Model dims (match reference)
#define B_  8
#define T_  256
#define D_  1024
#define L_  12
#define F_  4096
#define V_  32000
#define M_  (B_ * T_)   // 2048 token rows

// ---------------------------------------------------------------------------
// Types
// ---------------------------------------------------------------------------
typedef unsigned short bf16s;                                     // bf16 bits in memory
typedef __attribute__((ext_vector_type(16))) __bf16 v16bf;        // WMMA A/B operand
typedef __attribute__((ext_vector_type(8)))  float  v8f;          // WMMA C/D operand

__device__ __forceinline__ bf16s f32_to_bf16(float f) {
  unsigned int u = __float_as_uint(f);
  u += 0x7fffu + ((u >> 16) & 1u);     // round-to-nearest-even
  return (bf16s)(u >> 16);
}

union FragAB { float4 q[2]; v16bf v; };     // 32 bytes = 16 bf16 = 8 VGPRs
union Pack8  { __bf16 h[8]; float4 v; };    // 8 bf16 packed for one 16B LDS store

// ---------------------------------------------------------------------------
// bf16 WMMA GEMM:  C[M,N] = A[M,K](bf16) * W[N,K](f32, used transposed)
// block tile 128x128, 8 waves (wave32), wave tile 32x64 (2x4 WMMA 16x16x32)
// A operand staged via CDNA5 async global->LDS DMA (ASYNCcnt),
// B operand staged via VGPRs with in-register f32->bf16 conversion.
// MODE: 0 = plain, 1 = sigmoid(C), 2 = C = RES + acc (residual add)
// ---------------------------------------------------------------------------
#define BM   128
#define BN   128
#define BK   32
#define LDSW 40   // bf16 elements per LDS row (32 + 8 pad)

template<int MODE>
__global__ __launch_bounds__(256)
void gemm_bf16(const bf16s* __restrict__ A, const float* __restrict__ Bw,
               const float* __restrict__ RES, float* __restrict__ C,
               int M, int N, int K)
{
  __shared__ __align__(16) bf16s As[BM * LDSW];
  __shared__ __align__(16) bf16s Bs[BN * LDSW];

  const int tid  = threadIdx.x;
  const int lane = tid & 31;
  const int wid  = tid >> 5;     // 0..7
  const int wm   = wid & 3;      // 4 wave-rows  -> 32 rows each
  const int wn   = wid >> 2;     // 2 wave-cols  -> 64 cols each
  const int bm   = blockIdx.y * BM;
  const int bn   = blockIdx.x * BN;
  const int mr   = lane & 15;
  const int hi   = lane >> 4;    // half-wave select (ISA 16-bit operand layout)

  v8f acc[2][4];
#pragma unroll
  for (int i = 0; i < 2; ++i)
#pragma unroll
    for (int j = 0; j < 4; ++j)
#pragma unroll
      for (int r = 0; r < 8; ++r) acc[i][j][r] = 0.0f;

  // global->LDS staging assignments (256 threads)
  const int arow = tid >> 1;            // 0..127, 16 bf16 (32 B) each
  const int acol = (tid & 1) << 4;      // 0 or 16
  const int brow = tid >> 1;            // 0..127, 16 f32 each
  const int bcol = (tid & 1) << 4;      // 0 or 16

  // LDS byte address for async DMA: generic-pointer low 32 bits == LDS offset
  const unsigned lds_a  = (unsigned)(uintptr_t)&As[arow * LDSW + acol];
  const bf16s*  agp = A  + (size_t)(bm + arow) * K + acol;
  const float*  bgp = Bw + (size_t)(bn + brow) * K + bcol;
  bf16s*        bsp = &Bs[brow * LDSW + bcol];

  for (int k0 = 0; k0 < K; k0 += BK) {
    // ---- stage A tile 128x32 bf16 via async global->LDS (2 x B128 per lane)
    {
      unsigned long long ga = (unsigned long long)(uintptr_t)(agp + k0);
      asm volatile(
          "global_load_async_to_lds_b128 %0, %1, off\n\t"
          "global_load_async_to_lds_b128 %0, %1, off offset:16"
          :: "v"(lds_a), "v"(ga) : "memory");
    }
    // ---- stage B tile 128x32: f32 weights -> bf16 in-register -> LDS
    {
      const float* src = bgp + k0;
      float4 f0 = ((const float4*)src)[0];
      float4 f1 = ((const float4*)src)[1];
      float4 f2 = ((const float4*)src)[2];
      float4 f3 = ((const float4*)src)[3];
      Pack8 u0, u1;
      u0.h[0] = (__bf16)f0.x; u0.h[1] = (__bf16)f0.y;
      u0.h[2] = (__bf16)f0.z; u0.h[3] = (__bf16)f0.w;
      u0.h[4] = (__bf16)f1.x; u0.h[5] = (__bf16)f1.y;
      u0.h[6] = (__bf16)f1.z; u0.h[7] = (__bf16)f1.w;
      u1.h[0] = (__bf16)f2.x; u1.h[1] = (__bf16)f2.y;
      u1.h[2] = (__bf16)f2.z; u1.h[3] = (__bf16)f2.w;
      u1.h[4] = (__bf16)f3.x; u1.h[5] = (__bf16)f3.y;
      u1.h[6] = (__bf16)f3.z; u1.h[7] = (__bf16)f3.w;
      *((float4*)bsp)       = u0.v;
      *((float4*)(bsp + 8)) = u1.v;
      if (k0 + BK < K) __builtin_prefetch(src + BK, 0, 0);
    }
    // own async copies complete -> then workgroup barrier
    asm volatile("s_wait_asynccnt 0x0" ::: "memory");
    __syncthreads();

    // ---- fragments per ISA 16-bit layouts
    FragAB afrag[2], bfrag[4];
#pragma unroll
    for (int tm = 0; tm < 2; ++tm) {
      // A 16x32: lane<16 -> K[0..7],K[16..23]; lane>=16 -> K[8..15],K[24..31]
      const bf16s* p = &As[(wm * 32 + tm * 16 + mr) * LDSW + hi * 8];
      afrag[tm].q[0] = *(const float4*)p;
      afrag[tm].q[1] = *(const float4*)(p + 16);
    }
#pragma unroll
    for (int tn = 0; tn < 4; ++tn) {
      // B 32x16: lane<16 holds col=lane, K[0..7],K[16..23]; lane>=16 holds K[8..15],K[24..31]
      const bf16s* p = &Bs[(wn * 64 + tn * 16 + mr) * LDSW + hi * 8];
      bfrag[tn].q[0] = *(const float4*)p;
      bfrag[tn].q[1] = *(const float4*)(p + 16);
    }

#pragma unroll
    for (int tm = 0; tm < 2; ++tm)
#pragma unroll
      for (int tn = 0; tn < 4; ++tn)
        acc[tm][tn] = __builtin_amdgcn_wmma_f32_16x16x32_bf16(
            false, afrag[tm].v, false, bfrag[tn].v,
            (short)0, acc[tm][tn], false, false);

    __syncthreads();
  }

  // ---- epilogue (C/D layout: lane<16 -> M=r, lane>=16 -> M=8+r; N=lane&15)
#pragma unroll
  for (int tm = 0; tm < 2; ++tm)
#pragma unroll
    for (int tn = 0; tn < 4; ++tn) {
      const int row0 = bm + wm * 32 + tm * 16 + hi * 8;
      const int col  = bn + wn * 64 + tn * 16 + mr;
#pragma unroll
      for (int r = 0; r < 8; ++r) {
        size_t idx = (size_t)(row0 + r) * N + col;
        float v = acc[tm][tn][r];
        if (MODE == 1) v = 1.0f / (1.0f + __expf(-v));
        if (MODE == 2) v += RES[idx];
        C[idx] = v;
      }
    }
}

// ---------------------------------------------------------------------------
// Row kernels: one block (256 threads) per token row of D_=1024
// ---------------------------------------------------------------------------
__global__ __launch_bounds__(256)
void embed_rms_kernel(const int* __restrict__ x, const float* __restrict__ embed,
                      const float* __restrict__ w, float* __restrict__ h)
{
  __shared__ float red[256];
  const int row = blockIdx.x;
  const int tid = threadIdx.x;
  const float* e = embed + (size_t)x[row] * D_;
  float vals[D_ / 256];
  float s = 0.f;
#pragma unroll
  for (int i = 0; i < D_ / 256; ++i) { float v = e[tid + i * 256]; vals[i] = v; s += v * v; }
  red[tid] = s; __syncthreads();
  for (int o = 128; o > 0; o >>= 1) { if (tid < o) red[tid] += red[tid + o]; __syncthreads(); }
  const float rs = rsqrtf(red[0] / (float)D_ + 1e-6f);
#pragma unroll
  for (int i = 0; i < D_ / 256; ++i) {
    int j = tid + i * 256;
    h[(size_t)row * D_ + j] = vals[i] * rs * w[j];
  }
}

__global__ __launch_bounds__(256)
void rms_bf16_kernel(const float* __restrict__ x, const float* __restrict__ w,
                     bf16s* __restrict__ out)
{
  __shared__ float red[256];
  const int row = blockIdx.x;
  const int tid = threadIdx.x;
  const float* p = x + (size_t)row * D_;
  float vals[D_ / 256];
  float s = 0.f;
#pragma unroll
  for (int i = 0; i < D_ / 256; ++i) { float v = p[tid + i * 256]; vals[i] = v; s += v * v; }
  red[tid] = s; __syncthreads();
  for (int o = 128; o > 0; o >>= 1) { if (tid < o) red[tid] += red[tid + o]; __syncthreads(); }
  const float rs = rsqrtf(red[0] / (float)D_ + 1e-6f);
#pragma unroll
  for (int i = 0; i < D_ / 256; ++i) {
    int j = tid + i * 256;
    out[(size_t)row * D_ + j] = f32_to_bf16(vals[i] * rs * w[j]);
  }
}

__global__ __launch_bounds__(256)
void mul_rms_bf16_kernel(const float* __restrict__ a, const float* __restrict__ b,
                         const float* __restrict__ w, bf16s* __restrict__ out)
{
  __shared__ float red[256];
  const int row = blockIdx.x;
  const int tid = threadIdx.x;
  const size_t base = (size_t)row * D_;
  float vals[D_ / 256];
  float s = 0.f;
#pragma unroll
  for (int i = 0; i < D_ / 256; ++i) {
    int j = tid + i * 256;
    float v = a[base + j] * b[base + j];
    vals[i] = v; s += v * v;
  }
  red[tid] = s; __syncthreads();
  for (int o = 128; o > 0; o >>= 1) { if (tid < o) red[tid] += red[tid + o]; __syncthreads(); }
  const float rs = rsqrtf(red[0] / (float)D_ + 1e-6f);
#pragma unroll
  for (int i = 0; i < D_ / 256; ++i) {
    int j = tid + i * 256;
    out[base + j] = f32_to_bf16(vals[i] * rs * w[j]);
  }
}

// RWKV decay recurrence: s_t = sigmoid(decay_d) * s_{t-1} + k_t*v_t
__global__ __launch_bounds__(256)
void scan_kernel(const float* __restrict__ k, const float* __restrict__ v,
                 const float* __restrict__ decay, float* __restrict__ st)
{
  const int idx = blockIdx.x * blockDim.x + threadIdx.x;  // 0 .. B_*D_-1
  const int b = idx / D_;
  const int d = idx % D_;
  const float dec = 1.0f / (1.0f + __expf(-decay[d]));
  float s = 0.f;
  size_t base = (size_t)b * T_ * D_ + d;
  for (int t = 0; t < T_; ++t) {
    size_t o = base + (size_t)t * D_;
    s = dec * s + k[o] * v[o];
    st[o] = s;
  }
}

__global__ __launch_bounds__(256)
void silu_mul_kernel(const float* __restrict__ g1, const float* __restrict__ g2,
                     bf16s* __restrict__ out, int n)
{
  int i = blockIdx.x * blockDim.x + threadIdx.x;
  const int stride = gridDim.x * blockDim.x;
  for (; i < n; i += stride) {
    float a = g1[i];
    float s = a / (1.0f + __expf(-a));
    out[i] = f32_to_bf16(s * g2[i]);
  }
}

// ---------------------------------------------------------------------------
// Host orchestration
// ---------------------------------------------------------------------------
extern "C" void kernel_launch(void* const* d_in, const int* in_sizes, int n_in,
                              void* d_out, int out_size, void* d_ws, size_t ws_size,
                              hipStream_t stream) {
  (void)in_sizes; (void)n_in; (void)out_size; (void)ws_size;

  const int*   x      = (const int*)  d_in[0];
  const float* embed  = (const float*)d_in[1];
  const float* ln_in  = (const float*)d_in[2];
  const float* ln_out = (const float*)d_in[3];
  const float* ln1    = (const float*)d_in[4];
  const float* Wr     = (const float*)d_in[5];
  const float* Wk     = (const float*)d_in[6];
  const float* Wv     = (const float*)d_in[7];
  const float* Wo     = (const float*)d_in[8];
  const float* decay  = (const float*)d_in[9];
  const float* lnx    = (const float*)d_in[10];
  const float* ln2    = (const float*)d_in[11];
  const float* W1     = (const float*)d_in[12];
  const float* W2     = (const float*)d_in[13];
  const float* Wo2    = (const float*)d_in[14];
  float* logits = (float*)d_out;

  // workspace carve-out
  char* ws = (char*)d_ws;
  size_t off = 0;
  auto alloc = [&](size_t bytes) -> void* {
    off = (off + 255) & ~(size_t)255;
    void* p = ws + off;
    off += bytes;
    return p;
  };
  float* h   = (float*)alloc((size_t)M_ * D_ * 4);
  bf16s* xnb = (bf16s*)alloc((size_t)M_ * D_ * 2);
  float* rb  = (float*)alloc((size_t)M_ * D_ * 4);
  float* kb  = (float*)alloc((size_t)M_ * D_ * 4);
  float* vb  = (float*)alloc((size_t)M_ * D_ * 4);
  float* sb  = (float*)alloc((size_t)M_ * D_ * 4);
  bf16s* yb  = (bf16s*)alloc((size_t)M_ * D_ * 2);
  float* g1  = (float*)alloc((size_t)M_ * F_ * 4);
  float* g2  = (float*)alloc((size_t)M_ * F_ * 4);
  bf16s* gb  = (bf16s*)alloc((size_t)M_ * F_ * 2);

  const dim3 blk(256);
  const dim3 grid_rows(M_);                       // per-token reductions
  const dim3 g_DD(D_ / BN, M_ / BM);              // N=1024 GEMMs
  const dim3 g_DF(F_ / BN, M_ / BM);              // N=4096 GEMMs
  const dim3 g_V (V_ / BN, M_ / BM);              // logits GEMM

  // h = rms(embed[x], ln_in_w)
  embed_rms_kernel<<<grid_rows, blk, 0, stream>>>(x, embed, ln_in, h);

  for (int l = 0; l < L_; ++l) {
    const float* Wr_l  = Wr  + (size_t)l * D_ * D_;
    const float* Wk_l  = Wk  + (size_t)l * D_ * D_;
    const float* Wv_l  = Wv  + (size_t)l * D_ * D_;
    const float* Wo_l  = Wo  + (size_t)l * D_ * D_;
    const float* W1_l  = W1  + (size_t)l * F_ * D_;
    const float* W2_l  = W2  + (size_t)l * F_ * D_;
    const float* Wo2_l = Wo2 + (size_t)l * D_ * F_;

    // ---- TimeMix ----
    rms_bf16_kernel<<<grid_rows, blk, 0, stream>>>(h, ln1 + l * D_, xnb);
    gemm_bf16<1><<<g_DD, blk, 0, stream>>>(xnb, Wr_l, nullptr, rb, M_, D_, D_); // r = sigmoid
    gemm_bf16<0><<<g_DD, blk, 0, stream>>>(xnb, Wk_l, nullptr, kb, M_, D_, D_);
    gemm_bf16<0><<<g_DD, blk, 0, stream>>>(xnb, Wv_l, nullptr, vb, M_, D_, D_);
    scan_kernel<<<dim3((B_ * D_) / 256), blk, 0, stream>>>(kb, vb, decay + l * D_, sb);
    mul_rms_bf16_kernel<<<grid_rows, blk, 0, stream>>>(rb, sb, lnx + l * D_, yb);
    gemm_bf16<2><<<g_DD, blk, 0, stream>>>(yb, Wo_l, h, h, M_, D_, D_);         // h += y @ Wo^T

    // ---- ChannelMix ----
    rms_bf16_kernel<<<grid_rows, blk, 0, stream>>>(h, ln2 + l * D_, xnb);
    gemm_bf16<0><<<g_DF, blk, 0, stream>>>(xnb, W1_l, nullptr, g1, M_, F_, D_);
    gemm_bf16<0><<<g_DF, blk, 0, stream>>>(xnb, W2_l, nullptr, g2, M_, F_, D_);
    silu_mul_kernel<<<dim3(4096), blk, 0, stream>>>(g1, g2, gb, M_ * F_);
    gemm_bf16<2><<<g_DD, blk, 0, stream>>>(gb, Wo2_l, h, h, M_, D_, F_);        // h += g @ Wo2^T
  }

  // logits = rms(h, ln_out_w) @ embed^T
  rms_bf16_kernel<<<grid_rows, blk, 0, stream>>>(h, ln_out, xnb);
  gemm_bf16<0><<<g_V, blk, 0, stream>>>(xnb, embed, nullptr, logits, M_, V_, D_);
}
